// CausalAttention_25391846654288
// MI455X (gfx1250) — compile-verified, matrix operand order checked by reading, not run on hardware
//
#include <hip/hip_runtime.h>

#define SEQ  4096
#define DIN  1024
#define DOUT 1024
#define SROW 4100   // padded LDS row pitch for score strip (bank-conflict break)

typedef __attribute__((ext_vector_type(4)))  float        fvec4;
typedef __attribute__((ext_vector_type(8)))  float        v8f;
typedef __attribute__((ext_vector_type(4)))  unsigned int uvec4;
typedef __attribute__((ext_vector_type(8)))  unsigned int uvec8;
typedef __attribute__((ext_vector_type(16))) __bf16       v16bf;

__device__ __forceinline__ unsigned short f2bf(float f) {
  unsigned int u = __builtin_bit_cast(unsigned int, f);
  u = (u + 0x7FFFu + ((u >> 16) & 1u)) >> 16;   // round-to-nearest-even
  return (unsigned short)u;
}

__device__ __forceinline__ v16bf make_op(uvec4 a, uvec4 b) {
  uvec8 w;
  w[0]=a[0]; w[1]=a[1]; w[2]=a[2]; w[3]=a[3];
  w[4]=b[0]; w[5]=b[1]; w[6]=b[2]; w[7]=b[3];
  return __builtin_bit_cast(v16bf, w);
}

// 16x32 bf16 WMMA operand from a row-major bf16 matrix (leading dim ld).
// lane L: row = row0 + (L&15); K chunks at k0 + (L>>4)*8 and +16 (ISA 16-bit A layout).
__device__ __forceinline__ v16bf load_op_global(const unsigned short* __restrict__ base,
                                                int ld, int row0, int k0, int lane) {
  int r = row0 + (lane & 15);
  int c = k0 + ((lane >> 4) << 3);
  const unsigned short* p = base + (size_t)r * ld + c;
  uvec4 lo = *(const uvec4*)p;
  uvec4 hi = *(const uvec4*)(p + 16);
  return make_op(lo, hi);
}

__device__ __forceinline__ v16bf load_op_lds(const unsigned short* base,
                                             int ldr, int row0, int lane) {
  int r = row0 + (lane & 15);
  int c = (lane >> 4) << 3;
  const unsigned short* p = base + r * ldr + c;
  uvec4 lo = *(const uvec4*)p;
  uvec4 hi = *(const uvec4*)(p + 16);
  return make_op(lo, hi);
}

#define WMMA_BF16(a, b, c) \
  __builtin_amdgcn_wmma_f32_16x16x32_bf16(false, (a), false, (b), (short)0, (c), false, false)

// ---------------- kernel 1: C[4096,1024] = X[4096,1024] @ W[1024,1024] ----------------
// 8 waves (4 M x 2 N), block tile 128x64, k-step 32. STOREBF: bf16 (Q,K) vs f32 (V) output.
template <bool STOREBF>
__global__ void __launch_bounds__(256) proj_gemm_kernel(
    const float* __restrict__ X, const float* __restrict__ W,
    unsigned short* __restrict__ outBf, float* __restrict__ outF)
{
  __shared__ __align__(16) unsigned short As[128 * 40];
  __shared__ __align__(16) unsigned short Bs[64 * 40];
  const int t = threadIdx.x;
  const int wave = t >> 5, lane = t & 31;
  const int wm = wave & 3, wn = wave >> 2;
  const int m0 = blockIdx.y * 128, n0 = blockIdx.x * 64;

  v8f acc[2][2] = {};

  for (int kb = 0; kb < DIN; kb += 32) {
    // A tile: X[m0..+127][kb..+31] -> bf16 LDS (pitch 40)
#pragma unroll
    for (int i = 0; i < 4; i++) {
      int lin = i * 256 + t;            // 0..1023
      int row = lin >> 3;               // 0..127
      int cv  = (lin & 7) << 2;         // 0..28
      fvec4 v = *(const fvec4*)(X + (size_t)(m0 + row) * DIN + kb + cv);
      unsigned short* d = As + row * 40 + cv;
      d[0]=f2bf(v[0]); d[1]=f2bf(v[1]); d[2]=f2bf(v[2]); d[3]=f2bf(v[3]);
    }
    // B tile (transposed): Bs[n][k] = W[kb+k][n0+n]
#pragma unroll
    for (int i = 0; i < 2; i++) {
      int lin = i * 256 + t;            // 0..511
      int k  = lin >> 4;                // 0..31
      int nv = (lin & 15) << 2;         // 0..60
      fvec4 v = *(const fvec4*)(W + (size_t)(kb + k) * DOUT + n0 + nv);
      Bs[(nv + 0) * 40 + k] = f2bf(v[0]);
      Bs[(nv + 1) * 40 + k] = f2bf(v[1]);
      Bs[(nv + 2) * 40 + k] = f2bf(v[2]);
      Bs[(nv + 3) * 40 + k] = f2bf(v[3]);
    }
    __syncthreads();

    v16bf a0 = load_op_lds(As, 40, wm * 32 +  0, lane);
    v16bf a1 = load_op_lds(As, 40, wm * 32 + 16, lane);
    v16bf b0 = load_op_lds(Bs, 40, wn * 32 +  0, lane);
    v16bf b1 = load_op_lds(Bs, 40, wn * 32 + 16, lane);
    acc[0][0] = WMMA_BF16(a0, b0, acc[0][0]);
    acc[0][1] = WMMA_BF16(a0, b1, acc[0][1]);
    acc[1][0] = WMMA_BF16(a1, b0, acc[1][0]);
    acc[1][1] = WMMA_BF16(a1, b1, acc[1][1]);
    __syncthreads();
  }

#pragma unroll
  for (int mi = 0; mi < 2; mi++)
#pragma unroll
    for (int ni = 0; ni < 2; ni++)
#pragma unroll
      for (int r = 0; r < 8; r++) {
        int row = m0 + wm * 32 + mi * 16 + ((lane >> 4) << 3) + r;
        int col = n0 + wn * 32 + ni * 16 + (lane & 15);
        float v = acc[mi][ni][r];
        if (STOREBF) outBf[(size_t)row * DOUT + col] = f2bf(v);
        else         outF [(size_t)row * DOUT + col] = v;
      }
}

// ---------------- kernels 2a/2b/2c: in-place prefix sum of V rows -> Vcum ----------------
__global__ void __launch_bounds__(256) seg_scan_kernel(float* __restrict__ V,
                                                       float* __restrict__ segtot) {
  int col = blockIdx.x * 256 + threadIdx.x;
  int seg = blockIdx.y;
  float s = 0.f;
  size_t base = (size_t)seg * 256 * DOUT + col;
  for (int r = 0; r < 256; r++) {
    size_t idx = base + (size_t)r * DOUT;
    s += V[idx];
    V[idx] = s;
  }
  segtot[seg * DOUT + col] = s;
}

__global__ void __launch_bounds__(256) seg_off_kernel(float* __restrict__ segtot) {
  int col = blockIdx.x * 256 + threadIdx.x;
  float off = 0.f;
  for (int s = 0; s < 16; s++) {
    float v = segtot[s * DOUT + col];
    segtot[s * DOUT + col] = off;   // exclusive offsets
    off += v;
  }
}

__global__ void __launch_bounds__(256) seg_add_kernel(float* __restrict__ V,
                                                      const float* __restrict__ segtot) {
  int col = blockIdx.x * 256 + threadIdx.x;
  int seg = blockIdx.y;
  float off = segtot[seg * DOUT + col];
  size_t base = (size_t)seg * 256 * DOUT + col;
  for (int r = 0; r < 256; r++) V[base + (size_t)r * DOUT] += off;
}

// ---------------- kernel 3: scores + max + E=exp(s-m) + Z = sum (t+1) E ----------------
// One 16-row block per WG; 16xSROW fp32 score strip in dynamic LDS (256 KB of 320 KB WGP LDS).
__global__ void __launch_bounds__(256) scores_kernel(
    const unsigned short* __restrict__ Q, const unsigned short* __restrict__ K,
    unsigned short* __restrict__ P, float* __restrict__ Zw)
{
  extern __shared__ float sc[];   // [16][SROW]
  __shared__ float s_rmax[16];
  __shared__ float s_zw[16];
  const int t = threadIdx.x, wave = t >> 5, lane = t & 31;
  const int i0 = blockIdx.x * 16;
  const float scale = 0.03125f;   // 1/sqrt(1024)

  // phase 1: full 16x4096 score strip via WMMA (K rows are already the N x K layout -> no transpose)
  // Software-pipelined: loads for iteration ks+1 are issued before the WMMAs of
  // iteration ks, so the waitcnt pass can use partial s_wait_loadcnt and keep
  // the next 10 loads in flight behind the WMMA quartet.
  for (int it = 0; it < 8; it++) {
    int jt0 = (it * 8 + wave) * 4;            // 4 consecutive 16-wide j tiles per wave per pass
    v8f acc[4] = {};

    v16bf a = load_op_global(Q, DIN, i0, 0, lane);
    v16bf b[4];
#pragma unroll
    for (int jc = 0; jc < 4; jc++)
      b[jc] = load_op_global(K, DIN, (jt0 + jc) * 16, 0, lane);

    for (int ks = 0; ks < 31; ks++) {
      v16bf an = load_op_global(Q, DIN, i0, (ks + 1) * 32, lane);
      v16bf bn[4];
#pragma unroll
      for (int jc = 0; jc < 4; jc++)
        bn[jc] = load_op_global(K, DIN, (jt0 + jc) * 16, (ks + 1) * 32, lane);
#pragma unroll
      for (int jc = 0; jc < 4; jc++)
        acc[jc] = WMMA_BF16(a, b[jc], acc[jc]);
      a = an;
#pragma unroll
      for (int jc = 0; jc < 4; jc++) b[jc] = bn[jc];
    }
#pragma unroll
    for (int jc = 0; jc < 4; jc++)
      acc[jc] = WMMA_BF16(a, b[jc], acc[jc]);

#pragma unroll
    for (int jc = 0; jc < 4; jc++) {
      int colb = (jt0 + jc) * 16 + (lane & 15);
#pragma unroll
      for (int r = 0; r < 8; r++) {
        int row = ((lane >> 4) << 3) + r;
        sc[row * SROW + colb] = acc[jc][r] * scale;
      }
    }
  }
  __syncthreads();

  // phase 2: row max (wave w owns rows 2w, 2w+1; stride-32 lanes -> conflict-free)
  for (int rr = 0; rr < 2; rr++) {
    int row = wave * 2 + rr;
    float m = -3.0e38f;
    for (int k2 = 0; k2 < 128; k2++)
      m = fmaxf(m, sc[row * SROW + lane + 32 * k2]);
#pragma unroll
    for (int off = 16; off > 0; off >>= 1)
      m = fmaxf(m, __shfl_xor(m, off, 32));
    if (lane == 0) s_rmax[row] = m;
  }
  __syncthreads();

  // phase 3: Z = sum (t+1) * exp(s - m)  (softmax denominator cancels analytically)
  for (int rr = 0; rr < 2; rr++) {
    int row = wave * 2 + rr;
    float m = s_rmax[row];
    float z = 0.f;
    for (int k2 = 0; k2 < 128; k2++) {
      int c = lane + 32 * k2;
      z += __expf(sc[row * SROW + c] - m) * (float)(c + 1);
    }
#pragma unroll
    for (int off = 16; off > 0; off >>= 1)
      z += __shfl_xor(z, off, 32);
    if (lane == 0) s_zw[row] = z;
  }
  __syncthreads();
  if (t < 16) Zw[i0 + t] = s_zw[t];

  // phase 4: coalesced bf16 store of E = exp(s - m)
  for (int k2 = 0; k2 < 256; k2++) {
    int idx = k2 * 256 + t;
    int row = idx >> 12;
    int col = idx & 4095;
    float e = __expf(sc[row * SROW + col] - s_rmax[row]);
    P[(size_t)(i0 + row) * SEQ + col] = f2bf(e);
  }
}

// ---------------- kernel 4: Out = (E @ Vcum) / Z ----------------
__global__ void __launch_bounds__(256) out_gemm_kernel(
    const unsigned short* __restrict__ P, const float* __restrict__ Vc,
    const float* __restrict__ Zw, float* __restrict__ Out)
{
  __shared__ __align__(16) unsigned short Bs[64 * 40];
  const int t = threadIdx.x, wave = t >> 5, lane = t & 31;
  const int wm = wave & 3, wn = wave >> 2;
  const int m0 = blockIdx.y * 128, n0 = blockIdx.x * 64;
  v8f acc[2][2] = {};

  for (int kb = 0; kb < SEQ; kb += 32) {
    // B tile (transposed + bf16): Bs[n][k] = Vcum[kb+k][n0+n]
#pragma unroll
    for (int i = 0; i < 2; i++) {
      int lin = i * 256 + t;
      int k  = lin >> 4;
      int nv = (lin & 15) << 2;
      fvec4 v = *(const fvec4*)(Vc + (size_t)(kb + k) * DOUT + n0 + nv);
      Bs[(nv + 0) * 40 + k] = f2bf(v[0]);
      Bs[(nv + 1) * 40 + k] = f2bf(v[1]);
      Bs[(nv + 2) * 40 + k] = f2bf(v[2]);
      Bs[(nv + 3) * 40 + k] = f2bf(v[3]);
      if (kb + 32 < SEQ)
        __builtin_prefetch(Vc + (size_t)(kb + 32 + k) * DOUT + n0 + nv, 0, 1);
    }
    __syncthreads();

    v16bf a0 = load_op_global(P, SEQ, m0 + wm * 32 +  0, kb, lane);
    v16bf a1 = load_op_global(P, SEQ, m0 + wm * 32 + 16, kb, lane);
    v16bf b0 = load_op_lds(Bs, 40, wn * 32 +  0, lane);
    v16bf b1 = load_op_lds(Bs, 40, wn * 32 + 16, lane);
    acc[0][0] = WMMA_BF16(a0, b0, acc[0][0]);
    acc[0][1] = WMMA_BF16(a0, b1, acc[0][1]);
    acc[1][0] = WMMA_BF16(a1, b0, acc[1][0]);
    acc[1][1] = WMMA_BF16(a1, b1, acc[1][1]);
    __syncthreads();
  }

#pragma unroll
  for (int mi = 0; mi < 2; mi++)
#pragma unroll
    for (int ni = 0; ni < 2; ni++)
#pragma unroll
      for (int r = 0; r < 8; r++) {
        int row = m0 + wm * 32 + mi * 16 + ((lane >> 4) << 3) + r;
        int col = n0 + wn * 32 + ni * 16 + (lane & 15);
        Out[(size_t)row * DOUT + col] = acc[mi][ni][r] / Zw[row];
      }
}

// ---------------- host launcher ----------------
extern "C" void kernel_launch(void* const* d_in, const int* in_sizes, int n_in,
                              void* d_out, int out_size, void* d_ws, size_t ws_size,
                              hipStream_t stream) {
  const float* x  = (const float*)d_in[0];
  const float* wq = (const float*)d_in[1];
  const float* wk = (const float*)d_in[2];
  const float* wv = (const float*)d_in[3];
  float* out = (float*)d_out;

  char* ws = (char*)d_ws;
  unsigned short* Qbf = (unsigned short*)(ws);                      //  8 MB bf16
  unsigned short* Kbf = (unsigned short*)(ws + (size_t)( 8 << 20)); //  8 MB bf16
  float*          Vc  = (float*)         (ws + (size_t)(16 << 20)); // 16 MB f32 (V -> Vcum)
  unsigned short* P   = (unsigned short*)(ws + (size_t)(32 << 20)); // 32 MB bf16 (E)
  float*          Zw  = (float*)         (ws + (size_t)(64 << 20)); // 16 KB
  float*          seg = (float*)         (ws + (size_t)(64 << 20) + (1 << 16)); // 64 KB

  dim3 blk(256);
  dim3 gGemm(DOUT / 64, SEQ / 128);

  proj_gemm_kernel<true> <<<gGemm, blk, 0, stream>>>(x, wq, Qbf, nullptr);
  proj_gemm_kernel<true> <<<gGemm, blk, 0, stream>>>(x, wk, Kbf, nullptr);
  proj_gemm_kernel<false><<<gGemm, blk, 0, stream>>>(x, wv, nullptr, Vc);

  seg_scan_kernel<<<dim3(4, 16), blk, 0, stream>>>(Vc, seg);
  seg_off_kernel <<<dim3(4),     blk, 0, stream>>>(seg);
  seg_add_kernel <<<dim3(4, 16), blk, 0, stream>>>(Vc, seg);

  size_t smem = (size_t)16 * SROW * sizeof(float);  // 256.25 KB dynamic LDS
  scores_kernel<<<dim3(SEQ / 16), blk, smem, stream>>>(Qbf, Kbf, P, Zw);

  out_gemm_kernel<<<gGemm, blk, 0, stream>>>(P, Vc, Zw, out);
}
